// HeteroSAGE_41532333752931
// MI455X (gfx1250) — matrix operand order repaired and checked by reading, not compiled, not run on hardware
//
#include <hip/hip_runtime.h>
#include <hip/hip_bf16.h>

typedef __attribute__((ext_vector_type(16))) _Float16 v16h;
typedef __attribute__((ext_vector_type(8)))  float    v8f;

#define H 64

// ---------------------------------------------------------------------------
// WMMA helper: D = A(16x32 f16) * B(32x16 f16) + C(16x16 f32)
// ---------------------------------------------------------------------------
__device__ __forceinline__ v8f wmma_f16(v16h a, v16h b, v8f c) {
  return __builtin_amdgcn_wmma_f32_16x16x32_f16(
      /*neg_a=*/false, a, /*neg_b=*/false, b,
      /*c_mod=*/(short)0, c, /*reuse_a=*/false, /*reuse_b=*/false);
}

// A-operand loader (16x32 f16 tile). Per CDNA5 ISA 16-bit A layout:
// lane = half*16 + m ; VGPR j holds K = kbase + (j>>2)*16 + half*8 + (j&3)*2 (+0,+1)
// 'p' points at this lane's row (row-major, stride 64 f32). Scales on the fly.
__device__ __forceinline__ v16h load_a(const float* __restrict__ p, int kbase,
                                       int half, float scale) {
  v16h r;
#pragma unroll
  for (int j = 0; j < 8; ++j) {
    int k = kbase + ((j >> 2) << 4) + (half << 3) + ((j & 3) << 1);
    float2 v = *(const float2*)(p + k);
    r[2 * j]     = (_Float16)(v.x * scale);
    r[2 * j + 1] = (_Float16)(v.y * scale);
  }
  return r;
}

// B-operand loader (32x16 f16 tile). Same K order as load_a:
// lane holds column n = lane&15; VGPR j holds K = kbase + (j>>2)*16 + half*8 + (j&3)*2 (+0,+1).
// B(k,n) = W[n*64 + k]  (computes X @ W^T).  'p' = W + n*64.
__device__ __forceinline__ v16h load_b(const float* __restrict__ p, int kbase,
                                       int half) {
  v16h r;
#pragma unroll
  for (int j = 0; j < 8; ++j) {
    int k = kbase + ((j >> 2) << 4) + (half << 3) + ((j & 3) << 1);
    float2 v = *(const float2*)(p + k);
    r[2 * j]     = (_Float16)v.x;
    r[2 * j + 1] = (_Float16)v.y;
  }
  return r;
}

// ---------------------------------------------------------------------------
// zero fill (float4 granular; all our sizes are multiples of 4 floats)
// ---------------------------------------------------------------------------
__global__ void __launch_bounds__(256) zero_kernel(float4* __restrict__ p, int n4) {
  int i = blockIdx.x * 256 + threadIdx.x;
  if (i < n4) p[i] = float4{0.f, 0.f, 0.f, 0.f};
}

// ---------------------------------------------------------------------------
// degree counts: one thread per edge, bumps both endpoints' counters
// ---------------------------------------------------------------------------
__global__ void __launch_bounds__(256)
count_kernel(const int* __restrict__ row0, const int* __restrict__ row1,
             float* __restrict__ cnt0, float* __restrict__ cnt1, int E) {
  int i = blockIdx.x * 256 + threadIdx.x;
  if (i >= E) return;
  atomicAdd(&cnt0[row0[i]], 1.0f);
  atomicAdd(&cnt1[row1[i]], 1.0f);
}

// ---------------------------------------------------------------------------
// edge scatter: one wave32 per edge; 64 consecutive floats gathered from the
// source row (two fully-coalesced 128B transactions) and atomically added to
// the destination row. sum[] (25.6MB) lives in L2 (192MB), so the f32 atomics
// resolve on-chip.
// ---------------------------------------------------------------------------
__global__ void __launch_bounds__(256)
scatter_kernel(const int* __restrict__ src, const int* __restrict__ dst,
               const float* __restrict__ x, float* __restrict__ sum, int E) {
  int gw   = (blockIdx.x * 256 + threadIdx.x) >> 5;
  int lane = threadIdx.x & 31;
  if (gw >= E) return;
  int s = src[gw];
  int d = dst[gw];
  const float* xr = x + (size_t)s * H;
  float*       sr = sum + (size_t)d * H;
  float v0 = xr[lane];
  float v1 = xr[lane + 32];
  atomicAdd(sr + lane, v0);
  atomicAdd(sr + lane + 32, v1);
}

// ---------------------------------------------------------------------------
// Fused SAGE GEMM:
//   out[m, :] (op)= relu?( sum[m,:]*(1/max(cnt[m],1)) @ B1^T + bias
//                          + (A2 ? A2[m,:] @ B2^T : 0) + (accum ? out[m,:] : 0) )
// One wave per 16-row tile, 4 N-tiles of 16 cols, K=64 => 2 WMMAs per term.
// All control flow is wave-uniform so EXEC is all-1s at every WMMA.
// ---------------------------------------------------------------------------
__global__ void __launch_bounds__(256)
sage_gemm(const float* __restrict__ A1, const float* __restrict__ cnt,
          const float* __restrict__ B1, const float* __restrict__ A2,
          const float* __restrict__ B2, const float* __restrict__ bias,
          float* __restrict__ out, int nrows, int accum, int relu) {
  const int lane  = threadIdx.x & 31;
  const int wave  = threadIdx.x >> 5;
  const int mbase = (blockIdx.x * 8 + wave) * 16;
  if (mbase >= nrows) return;                 // wave-uniform: EXEC stays full

  const int m    = lane & 15;
  const int half = lane >> 4;
  const int node = mbase + m;

  float scale1 = 1.0f;
  if (cnt) scale1 = 1.0f / fmaxf(cnt[node], 1.0f);

  const float* a1row = A1 + (size_t)node * H;
  v16h a1lo = load_a(a1row, 0,  half, scale1);
  v16h a1hi = load_a(a1row, 32, half, scale1);

  v16h a2lo, a2hi;
  if (A2) {
    const float* a2row = A2 + (size_t)node * H;
    a2lo = load_a(a2row, 0,  half, 1.0f);
    a2hi = load_a(a2row, 32, half, 1.0f);
  }

#pragma unroll
  for (int nt = 0; nt < 4; ++nt) {
    const int nb   = nt * 16;
    const int ncol = nb + m;                  // this lane's output column

    v8f acc;
    const float bv = bias[ncol];
#pragma unroll
    for (int r = 0; r < 8; ++r) acc[r] = bv;

    const float* b1col = B1 + (size_t)ncol * H;
    v16h b1lo = load_b(b1col, 0,  half);
    v16h b1hi = load_b(b1col, 32, half);
    acc = wmma_f16(a1lo, b1lo, acc);
    acc = wmma_f16(a1hi, b1hi, acc);

    if (A2) {
      const float* b2col = B2 + (size_t)ncol * H;
      v16h b2lo = load_b(b2col, 0,  half);
      v16h b2hi = load_b(b2col, 32, half);
      acc = wmma_f16(a2lo, b2lo, acc);
      acc = wmma_f16(a2hi, b2hi, acc);
    }

    // C/D layout: lane n = lane&15; VGPR r holds row M = r + half*8
#pragma unroll
    for (int r = 0; r < 8; ++r) {
      const size_t row = (size_t)(mbase + r + half * 8);
      float v = acc[r];
      if (accum) v += out[row * H + ncol];
      if (relu)  v = fmaxf(v, 0.0f);
      out[row * H + ncol] = v;
    }
  }
}

// ---------------------------------------------------------------------------
// Host-side orchestration
// ---------------------------------------------------------------------------
extern "C" void kernel_launch(void* const* d_in, const int* in_sizes, int n_in,
                              void* d_out, int out_size, void* d_ws, size_t ws_size,
                              hipStream_t stream) {
  (void)n_in; (void)out_size; (void)ws_size;

  const int*   e_ub = (const int*)d_in[0];    // [2,E]
  const int*   e_um = (const int*)d_in[1];    // [2,E]
  const float* xu   = (const float*)d_in[2];  // [N,H]
  const float* xb   = (const float*)d_in[3];
  const float* xm   = (const float*)d_in[4];
  const float* Wl1  = (const float*)d_in[5];  // [4,H,H]
  const float* bl1  = (const float*)d_in[6];  // [4,H]
  const float* Wr1  = (const float*)d_in[7];
  const float* Wl2  = (const float*)d_in[8];
  const float* bl2  = (const float*)d_in[9];
  const float* Wr2  = (const float*)d_in[10];
  const float* linW = (const float*)d_in[11]; // [3,H,H]
  const float* linb = (const float*)d_in[12]; // [3,H]

  const int E = in_sizes[0] / 2;
  const int N = in_sizes[2] / H;
  const size_t NH = (size_t)N * H;

  const int* ub0 = e_ub;     const int* ub1 = e_ub + E;
  const int* um0 = e_um;     const int* um1 = e_um + E;

  float* ws    = (float*)d_ws;
  float* sumA  = ws;                 // [N,H] shared scatter target
  float* y_u   = ws + NH;            // layer-1 outputs (post-ReLU)
  float* y_b   = ws + 2 * NH;
  float* y_m   = ws + 3 * NH;
  float* cnt_b  = ws + 4 * NH;       // degree counts (layer-invariant)
  float* cnt_u1 = cnt_b + N;
  float* cnt_m  = cnt_b + 2 * N;
  float* cnt_u2 = cnt_b + 3 * N;

  float* z_u = (float*)d_out;        // layer-2 outputs live in d_out,
  float* z_b = (float*)d_out + NH;   // final linear runs in place
  float* z_m = (float*)d_out + 2 * NH;

  const int gemm_blocks = (N + 127) / 128;
  const int scat_blocks = (E * 32 + 255) / 256;
  const int edge_blocks = (E + 255) / 256;

  auto zero = [&](float* p, size_t nfloats) {
    int n4 = (int)(nfloats / 4);
    zero_kernel<<<(n4 + 255) / 256, 256, 0, stream>>>((float4*)p, n4);
  };

  auto relation = [&](const int* s, const int* d, const float* xsrc,
                      const float* cnt, const float* Wl, const float* bl,
                      const float* xdst, const float* Wr, float* out,
                      int accum, int relu) {
    zero(sumA, NH);
    scatter_kernel<<<scat_blocks, 256, 0, stream>>>(s, d, xsrc, sumA, E);
    sage_gemm<<<gemm_blocks, 256, 0, stream>>>(sumA, cnt, Wl, xdst, Wr, bl,
                                               out, N, accum, relu);
  };

  // --- degree counts (once; identical for both layers) ---
  zero(cnt_b, 4 * (size_t)N);
  count_kernel<<<edge_blocks, 256, 0, stream>>>(ub0, ub1, cnt_u1, cnt_b, E);
  count_kernel<<<edge_blocks, 256, 0, stream>>>(um0, um1, cnt_u2, cnt_m, E);

  // --- layer 1 (ReLU fused; user accumulates rel1 + rel3) ---
  relation(ub0, ub1, xu, cnt_b,  Wl1 + 0*H*H, bl1 + 0*H, xb, Wr1 + 0*H*H, y_b, 0, 1);
  relation(ub1, ub0, xb, cnt_u1, Wl1 + 1*H*H, bl1 + 1*H, xu, Wr1 + 1*H*H, y_u, 0, 0);
  relation(um0, um1, xu, cnt_m,  Wl1 + 2*H*H, bl1 + 2*H, xm, Wr1 + 2*H*H, y_m, 0, 1);
  relation(um1, um0, xm, cnt_u2, Wl1 + 3*H*H, bl1 + 3*H, xu, Wr1 + 3*H*H, y_u, 1, 1);

  // --- layer 2 (no ReLU) ---
  relation(ub0, ub1, y_u, cnt_b,  Wl2 + 0*H*H, bl2 + 0*H, y_b, Wr2 + 0*H*H, z_b, 0, 0);
  relation(ub1, ub0, y_b, cnt_u1, Wl2 + 1*H*H, bl2 + 1*H, y_u, Wr2 + 1*H*H, z_u, 0, 0);
  relation(um0, um1, y_u, cnt_m,  Wl2 + 2*H*H, bl2 + 2*H, y_m, Wr2 + 2*H*H, z_m, 0, 0);
  relation(um1, um0, y_m, cnt_u2, Wl2 + 3*H*H, bl2 + 3*H, y_u, Wr2 + 3*H*H, z_u, 1, 0);

  // --- final per-type linear + ReLU, in place on d_out ---
  // (each wave loads its whole 16x64 A-tile into registers before storing;
  //  tiles are disjoint, so in-place is race-free)
  sage_gemm<<<gemm_blocks, 256, 0, stream>>>(z_u, nullptr, linW + 0*H*H, nullptr,
                                             nullptr, linb + 0*H, z_u, N, 0, 1);
  sage_gemm<<<gemm_blocks, 256, 0, stream>>>(z_b, nullptr, linW + 1*H*H, nullptr,
                                             nullptr, linb + 1*H, z_b, N, 0, 1);
  sage_gemm<<<gemm_blocks, 256, 0, stream>>>(z_m, nullptr, linW + 2*H*H, nullptr,
                                             nullptr, linb + 2*H, z_m, N, 0, 1);
}
